// Network_31379031065030
// MI455X (gfx1250) — matrix operand order repaired and checked by reading, not compiled, not run on hardware
//
#include <hip/hip_runtime.h>
#include <math.h>

#define NN 6000
#define NE 120000
#define NG 8

#define INVN 0.22360679774997896f   // 1/sqrt(20)
#define RS3  0.5773502691896258f    // 1/sqrt(3)
#define RS2  0.7071067811865476f    // 1/sqrt(2)

typedef __bf16 bhalf;
typedef bhalf bh16 __attribute__((ext_vector_type(16)));
typedef float f32x8 __attribute__((ext_vector_type(8)));
typedef unsigned int u32x4 __attribute__((ext_vector_type(4)));

union BFrag { u32x4 q[2]; bh16 v; };

__device__ __forceinline__ unsigned short f2bf(float f) {
    unsigned u = __float_as_uint(f);
    unsigned r = u + 0x7FFFu + ((u >> 16) & 1u);
    return (unsigned short)(r >> 16);
}
// fast swish / sigmoid: v_exp + v_rcp (no IEEE division Newton chains)
__device__ __forceinline__ float swishf(float x) {
    return x * __builtin_amdgcn_rcpf(1.f + __expf(-x));
}
__device__ __forceinline__ float sigmf(float x) {
    return __builtin_amdgcn_rcpf(1.f + __expf(-x));
}

// A fragment: 16x32 bf16, row-major source, per-lane 16 consecutive K (two 16B chunks)
__device__ __forceinline__ bh16 mk_a(const unsigned short* p) {
    BFrag t;
    t.q[0] = *(const u32x4*)p;
    t.q[1] = *(const u32x4*)(p + 8);
    return t.v;
}
// B fragment: 32x16 bf16 from row-major W[N][K]; per-lane 16 consecutive K (two 16B chunks)
__device__ __forceinline__ bh16 mk_b(const unsigned short* p) {
    BFrag t;
    t.q[0] = *(const u32x4*)p;
    t.q[1] = *(const u32x4*)(p + 8);
    return t.v;
}

// ---------------------------------------------------------------------------
// Fused 3-layer radial MLP: out = (swish(swish(rad@W0^T/√50)@W1^T/16)@W2^T/16)*c
// 64-edge tiles, 4 waves x 16 rows, bf16 WMMA, f32 accumulate.
// Rad tile staged via async global->LDS DMA (ASYNCcnt).
// ---------------------------------------------------------------------------
__global__ __launch_bounds__(128) void k_radial_mlp(
    const unsigned short* __restrict__ radbf,   // [NE,64] bf16 (cols 50..63 zero)
    const float* __restrict__ cbuf,             // [NE] cutoff
    const unsigned short* __restrict__ W0,      // [256,64] bf16
    const unsigned short* __restrict__ W1,      // [256,256] bf16
    const unsigned short* __restrict__ W2,      // [NP,256] bf16 (rows >= wn zero)
    float* __restrict__ wout,                   // [chunk,NP]
    int ebase, int ecount, int NP)
{
    __shared__ unsigned short sH1[64 * 256];    // 32 KB
    __shared__ unsigned short sB[64 * 256];     // 32 KB: rad tile, then H2

    const int tid   = threadIdx.x;
    const int lane  = tid & 31;
    const int wave  = tid >> 5;
    const int tile0 = blockIdx.x * 64;

    __builtin_prefetch(W1, 0, 3);
    __builtin_prefetch(W2, 0, 3);

    // stage rad tile (64 rows x 64 bf16) into sB via async LDS DMA
    for (int i = tid; i < 64 * 8; i += 128) {
        int row = i >> 3, c8 = (i & 7) << 3;
        unsigned short* dstp = sB + row * 64 + c8;
        if (tile0 + row < ecount) {
            const void* g = (const void*)(radbf + (size_t)(ebase + tile0 + row) * 64 + c8);
            unsigned ldsoff = (unsigned)(uintptr_t)(void*)dstp;  // low 32 bits = LDS offset
            asm volatile("global_load_async_to_lds_b128 %0, %1, off"
                         :: "v"(ldsoff), "v"(g) : "memory");
        } else {
            *(u32x4*)dstp = (u32x4){0u, 0u, 0u, 0u};
        }
    }
    asm volatile("s_wait_asynccnt 0x0" ::: "memory");
    __syncthreads();

    const int mbase = wave << 4;
    const int n     = lane & 15;
    const int half  = lane >> 4;
    const int kg    = half;
    const int m     = lane & 15;

    // ---- stage 1: H1 = swish(rad @ W0^T / sqrt(50)) ----
    {
        const float s0 = 0.14142135623730950f;  // 1/sqrt(50)
        const unsigned short* arow = sB + (mbase + m) * 64;
        for (int nt = 0; nt < 16; ++nt) {
            f32x8 acc;
            #pragma unroll
            for (int r = 0; r < 8; ++r) acc[r] = 0.f;
            #pragma unroll
            for (int kt = 0; kt < 2; ++kt) {
                bh16 a = mk_a(arow + kt * 32 + kg * 16);
                bh16 b = mk_b(W0 + (size_t)((nt << 4) + n) * 64 + kt * 32 + kg * 16);
                acc = __builtin_amdgcn_wmma_f32_16x16x32_bf16(false, a, false, b,
                                                              (short)0, acc, false, false);
            }
            #pragma unroll
            for (int r = 0; r < 8; ++r)
                sH1[(mbase + r + (half << 3)) * 256 + (nt << 4) + n] = f2bf(swishf(acc[r] * s0));
        }
    }
    __syncthreads();

    // ---- stage 2: H2 = swish(H1 @ W1^T / 16) ----
    {
        const float s1 = 0.0625f;  // 1/sqrt(256)
        const unsigned short* arow = sH1 + (mbase + m) * 256;
        for (int nt = 0; nt < 16; ++nt) {
            f32x8 acc;
            #pragma unroll
            for (int r = 0; r < 8; ++r) acc[r] = 0.f;
            #pragma unroll
            for (int kt = 0; kt < 8; ++kt) {
                bh16 a = mk_a(arow + kt * 32 + kg * 16);
                bh16 b = mk_b(W1 + (size_t)((nt << 4) + n) * 256 + kt * 32 + kg * 16);
                acc = __builtin_amdgcn_wmma_f32_16x16x32_bf16(false, a, false, b,
                                                              (short)0, acc, false, false);
            }
            #pragma unroll
            for (int r = 0; r < 8; ++r)
                sB[(mbase + r + (half << 3)) * 256 + (nt << 4) + n] = f2bf(swishf(acc[r] * s1));
        }
    }
    __syncthreads();

    // ---- stage 3: out = (H2 @ W2^T / 16) * c[e] ----
    {
        const float s1 = 0.0625f;
        const unsigned short* arow = sB + (mbase + m) * 256;
        const int ntn = NP >> 4;
        for (int nt = 0; nt < ntn; ++nt) {
            f32x8 acc;
            #pragma unroll
            for (int r = 0; r < 8; ++r) acc[r] = 0.f;
            #pragma unroll
            for (int kt = 0; kt < 8; ++kt) {
                bh16 a = mk_a(arow + kt * 32 + kg * 16);
                bh16 b = mk_b(W2 + (size_t)((nt << 4) + n) * 256 + kt * 32 + kg * 16);
                acc = __builtin_amdgcn_wmma_f32_16x16x32_bf16(false, a, false, b,
                                                              (short)0, acc, false, false);
            }
            #pragma unroll
            for (int r = 0; r < 8; ++r) {
                int row = mbase + r + (half << 3);
                if (tile0 + row < ecount) {
                    float cw = cbuf[ebase + tile0 + row];
                    wout[(size_t)(tile0 + row) * NP + (nt << 4) + n] = acc[r] * s1 * cw;
                }
            }
        }
    }
}

// ---------------------------------------------------------------------------
// Small utility kernels
// ---------------------------------------------------------------------------
__global__ __launch_bounds__(256) void k_zero(float* p, int nel) {
    int i = blockIdx.x * 256 + threadIdx.x;
    if (i < nel) p[i] = 0.f;
}

__global__ __launch_bounds__(256) void k_w2bf(const float* __restrict__ W,
                                              unsigned short* __restrict__ Wb,
                                              int rows, int cols, int colsPad, int rowsPad) {
    int i = blockIdx.x * 256 + threadIdx.x;
    int tot = rowsPad * colsPad;
    if (i >= tot) return;
    int r = i / colsPad, c = i % colsPad;
    float v = (r < rows && c < cols) ? W[r * cols + c] : 0.f;
    Wb[i] = f2bf(v);
}

__global__ __launch_bounds__(256) void k_emb(const float* __restrict__ emb,
                                             const int* __restrict__ z,
                                             float* __restrict__ hs) {
    int i = blockIdx.x * 256 + threadIdx.x;
    if (i >= NN * 128) return;
    int nd = i >> 7, u = i & 127;
    hs[i] = emb[z[nd] * 128 + u];
}

__global__ __launch_bounds__(128) void k_edge_geom(const float* __restrict__ pos,
                                                   const int* __restrict__ src,
                                                   const int* __restrict__ dst,
                                                   unsigned short* __restrict__ radbf,
                                                   float* __restrict__ cbuf,
                                                   float* __restrict__ shv) {
    int e = blockIdx.x * 128 + threadIdx.x;
    if (e >= NE) return;
    int s = src[e], d = dst[e];
    float vx = pos[3 * s] - pos[3 * d];
    float vy = pos[3 * s + 1] - pos[3 * d + 1];
    float vz = pos[3 * s + 2] - pos[3 * d + 2];
    float dd = sqrtf(vx * vx + vy * vy + vz * vz + 1e-12f);
    float f = 1.7320508075688772f * INVN * __builtin_amdgcn_rcpf(dd);
    shv[3 * e] = f * vx; shv[3 * e + 1] = f * vy; shv[3 * e + 2] = f * vz;
    cbuf[e] = 0.5f * (__cosf(0.31415926535897931f * dd) + 1.f);
    unsigned short* rp = radbf + (size_t)e * 64;
    #pragma unroll 8
    for (int j = 0; j < 64; ++j) {
        float r = 0.f;
        if (j < 50) { float c = dd - (10.f / 49.f) * j; r = __expf(-12.5f * c * c); }
        rp[j] = f2bf(r);
    }
}

// y[n,o,c] (+)= scale * sum_u W[o,u] x[n,u,c]
__global__ __launch_bounds__(256) void k_linear(const float* __restrict__ W,
                                                const float* __restrict__ x,
                                                float* __restrict__ y,
                                                int mi, int mo, int nc, float scale, int accum) {
    int i = blockIdx.x * 256 + threadIdx.x;
    int tot = NN * mo * nc;
    if (i >= tot) return;
    int c = i % nc; int t = i / nc; int o = t % mo; int nd = t / mo;
    const float* xr = x + (size_t)nd * mi * nc + c;
    const float* wr = W + (size_t)o * mi;
    float a = 0.f;
    for (int u = 0; u < mi; ++u) a += wr[u] * xr[(size_t)u * nc];
    a *= scale;
    float* yp = y + ((size_t)nd * mo + o) * nc + c;
    if (accum) *yp += a; else *yp = a;
}

// ---------------------------------------------------------------------------
// Tensor-product + scatter kernels (one per conv, offsets from conv_struct)
// ---------------------------------------------------------------------------
__global__ __launch_bounds__(256) void k_tp0(const float* __restrict__ w, int NP,
        const float* __restrict__ xs, const int* __restrict__ src, const int* __restrict__ dst,
        const float* __restrict__ shv, float* __restrict__ ags, float* __restrict__ agv,
        int ebase, int ecount) {
    int idx = blockIdx.x * 256 + threadIdx.x;
    if (idx >= (ecount << 7)) return;
    int er = idx >> 7, u = idx & 127;
    int e = ebase + er;
    int s = src[e], d = dst[e];
    const float* we = w + (size_t)er * NP;
    float xj = xs[(size_t)s * 128 + u];
    atomicAdd(ags + (size_t)d * 128 + u, we[u] * xj * INVN);
    float wv = we[128 + u] * xj;
    float h0 = shv[3 * e], h1 = shv[3 * e + 1], h2 = shv[3 * e + 2];
    float* av = agv + ((size_t)d * 128 + u) * 3;
    atomicAdd(av, wv * h0); atomicAdd(av + 1, wv * h1); atomicAdd(av + 2, wv * h2);
}

__global__ __launch_bounds__(256) void k_tp1(const float* __restrict__ w, int NP,
        const float* __restrict__ xs, const float* __restrict__ xvm,
        const int* __restrict__ src, const int* __restrict__ dst, const float* __restrict__ shv,
        float* __restrict__ ags, float* __restrict__ agvm, float* __restrict__ agvp,
        int ebase, int ecount) {
    int idx = blockIdx.x * 256 + threadIdx.x;
    if (idx >= (ecount << 7)) return;
    int er = idx >> 7, u = idx & 127;
    int e = ebase + er;
    int s = src[e], d = dst[e];
    const float* we = w + (size_t)er * NP;
    float h0 = shv[3 * e], h1 = shv[3 * e + 1], h2 = shv[3 * e + 2];
    float xj = xs[(size_t)s * 128 + u];
    atomicAdd(ags + (size_t)d * 140 + u, we[u] * xj * INVN);
    float wv = we[128 + u] * xj;
    float* av = agvm + ((size_t)d * 140 + u) * 3;
    atomicAdd(av, wv * h0); atomicAdd(av + 1, wv * h1); atomicAdd(av + 2, wv * h2);
    if (u < 12) {
        const float* xv = xvm + ((size_t)s * 12 + u) * 3;
        float vx = xv[0], vy = xv[1], vz = xv[2];
        float w2 = we[256 + u] * INVN;
        float* a2 = agvm + ((size_t)d * 140 + 128 + u) * 3;
        atomicAdd(a2, w2 * vx); atomicAdd(a2 + 1, w2 * vy); atomicAdd(a2 + 2, w2 * vz);
        float dt = (vx * h0 + vy * h1 + vz * h2) * RS3;
        atomicAdd(ags + (size_t)d * 140 + 128 + u, we[268 + u] * dt);
        float cx = (vy * h2 - vz * h1) * RS2;
        float cy = (vz * h0 - vx * h2) * RS2;
        float cz = (vx * h1 - vy * h0) * RS2;
        float w4 = we[280 + u];
        float* a4 = agvp + ((size_t)d * 12 + u) * 3;
        atomicAdd(a4, w4 * cx); atomicAdd(a4 + 1, w4 * cy); atomicAdd(a4 + 2, w4 * cz);
    }
}

__global__ __launch_bounds__(256) void k_tp2(const float* __restrict__ w, int NP,
        const float* __restrict__ xs, const float* __restrict__ xvm, const float* __restrict__ xvp,
        const int* __restrict__ src, const int* __restrict__ dst, const float* __restrict__ shv,
        float* __restrict__ ags, float* __restrict__ agsm,
        float* __restrict__ agvm, float* __restrict__ agvp, int ebase, int ecount) {
    int idx = blockIdx.x * 256 + threadIdx.x;
    if (idx >= (ecount << 7)) return;
    int er = idx >> 7, u = idx & 127;
    int e = ebase + er;
    int s = src[e], d = dst[e];
    const float* we = w + (size_t)er * NP;
    float h0 = shv[3 * e], h1 = shv[3 * e + 1], h2 = shv[3 * e + 2];
    float xj = xs[(size_t)s * 128 + u];
    atomicAdd(ags + (size_t)d * 140 + u, we[u] * xj * INVN);
    float wv = we[128 + u] * xj;
    float* av = agvm + ((size_t)d * 152 + u) * 3;
    atomicAdd(av, wv * h0); atomicAdd(av + 1, wv * h1); atomicAdd(av + 2, wv * h2);
    if (u < 12) {
        const float* xm = xvm + ((size_t)s * 12 + u) * 3;
        const float* xp = xvp + ((size_t)s * 12 + u) * 3;
        float mx = xm[0], my = xm[1], mz = xm[2];
        float px = xp[0], py = xp[1], pz = xp[2];
        // I2: (1,-1)x(0,1)->(1,-1) at agvm[128+u]
        float w2 = we[256 + u] * INVN;
        float* a2 = agvm + ((size_t)d * 152 + 128 + u) * 3;
        atomicAdd(a2, w2 * mx); atomicAdd(a2 + 1, w2 * my); atomicAdd(a2 + 2, w2 * mz);
        // I3: dot -> ags[128+u]
        atomicAdd(ags + (size_t)d * 140 + 128 + u, we[268 + u] * (mx * h0 + my * h1 + mz * h2) * RS3);
        // I4: cross(xm,sh) -> agvp[u]
        float w4 = we[280 + u];
        float* a4 = agvp + ((size_t)d * 24 + u) * 3;
        atomicAdd(a4,     w4 * (my * h2 - mz * h1) * RS2);
        atomicAdd(a4 + 1, w4 * (mz * h0 - mx * h2) * RS2);
        atomicAdd(a4 + 2, w4 * (mx * h1 - my * h0) * RS2);
        // I5: (1,1)x(0,1)->(1,1) at agvp[12+u]
        float w5 = we[292 + u] * INVN;
        float* a5 = agvp + ((size_t)d * 24 + 12 + u) * 3;
        atomicAdd(a5, w5 * px); atomicAdd(a5 + 1, w5 * py); atomicAdd(a5 + 2, w5 * pz);
        // I6: dot(xp,sh) -> agsm[u]
        atomicAdd(agsm + (size_t)d * 12 + u, we[304 + u] * (px * h0 + py * h1 + pz * h2) * RS3);
        // I7: cross(xp,sh) -> agvm[140+u]
        float w7 = we[316 + u];
        float* a7 = agvm + ((size_t)d * 152 + 140 + u) * 3;
        atomicAdd(a7,     w7 * (py * h2 - pz * h1) * RS2);
        atomicAdd(a7 + 1, w7 * (pz * h0 - px * h2) * RS2);
        atomicAdd(a7 + 2, w7 * (px * h1 - py * h0) * RS2);
    }
}

__global__ __launch_bounds__(256) void k_tp3(const float* __restrict__ w, int NP,
        const float* __restrict__ xs, const float* __restrict__ xsm,
        const float* __restrict__ xvm, const float* __restrict__ xvp,
        const int* __restrict__ src, const int* __restrict__ dst, const float* __restrict__ shv,
        float* __restrict__ ags, float* __restrict__ agsm, int ebase, int ecount) {
    int idx = blockIdx.x * 256 + threadIdx.x;
    if (idx >= (ecount << 7)) return;
    int er = idx >> 7, u = idx & 127;
    int e = ebase + er;
    int s = src[e], d = dst[e];
    const float* we = w + (size_t)er * NP;
    float xj = xs[(size_t)s * 128 + u];
    float xq = xsm[(size_t)s * 128 + u];
    atomicAdd(ags + (size_t)d * 140 + u, we[u] * xj * INVN);
    atomicAdd(agsm + (size_t)d * 140 + u, we[128 + u] * xq * INVN);
    if (u < 12) {
        float h0 = shv[3 * e], h1 = shv[3 * e + 1], h2 = shv[3 * e + 2];
        const float* xm = xvm + ((size_t)s * 12 + u) * 3;
        const float* xp = xvp + ((size_t)s * 12 + u) * 3;
        float dm = (xm[0] * h0 + xm[1] * h1 + xm[2] * h2) * RS3;
        float dp = (xp[0] * h0 + xp[1] * h1 + xp[2] * h2) * RS3;
        atomicAdd(ags + (size_t)d * 140 + 128 + u, we[256 + u] * dm);
        atomicAdd(agsm + (size_t)d * 140 + 128 + u, we[268 + u] * dp);
    }
}

// ---------------------------------------------------------------------------
// Gating + readout
// ---------------------------------------------------------------------------
__global__ __launch_bounds__(256) void k_gate_s(const float* __restrict__ in, int sw,
                                                float* __restrict__ out) {
    int i = blockIdx.x * 256 + threadIdx.x;
    if (i >= NN * 128) return;
    int nd = i >> 7, u = i & 127;
    out[i] = swishf(in[(size_t)nd * sw + u]);
}
__global__ __launch_bounds__(256) void k_gate_t(const float* __restrict__ in, int sw,
                                                float* __restrict__ out) {
    int i = blockIdx.x * 256 + threadIdx.x;
    if (i >= NN * 128) return;
    int nd = i >> 7, u = i & 127;
    out[i] = tanhf(in[(size_t)nd * sw + u]);
}
__global__ __launch_bounds__(256) void k_gatev(const float* __restrict__ scal, int sw, int goff,
                                               const float* __restrict__ v, float* __restrict__ hv) {
    int i = blockIdx.x * 256 + threadIdx.x;
    if (i >= NN * 12 * 3) return;
    int t = i / 3; int u = t % 12; int nd = t / 12;
    float g = sigmf(scal[(size_t)nd * sw + goff + u]);
    hv[i] = v[i] * g;
}
__global__ __launch_bounds__(256) void k_readout(const float* __restrict__ fs,
                                                 const float* __restrict__ fsm,
                                                 const int* __restrict__ batch,
                                                 float* __restrict__ out) {
    int nd = blockIdx.x * 256 + threadIdx.x;
    if (nd >= NN) return;
    float v = fs[nd] + 0.5f * fsm[nd] * fsm[nd];
    atomicAdd(&out[batch[nd]], v);
}

// ---------------------------------------------------------------------------
static inline int cdiv(int a, int b) { return (a + b - 1) / b; }

extern "C" void kernel_launch(void* const* d_in, const int* in_sizes, int n_in,
                              void* d_out, int out_size, void* d_ws, size_t ws_size,
                              hipStream_t stream) {
    // --- input map (JAX pytree order: convs[fc,lin1,lin2,si sorted keys], embedding, z, pos, batch, src, dst)
    const float* FC0[4] = {(const float*)d_in[0], (const float*)d_in[7], (const float*)d_in[17], (const float*)d_in[30]};
    const float* FC1[4] = {(const float*)d_in[1], (const float*)d_in[8], (const float*)d_in[18], (const float*)d_in[31]};
    const float* FC2[4] = {(const float*)d_in[2], (const float*)d_in[9], (const float*)d_in[19], (const float*)d_in[32]};
    const int WN[4]  = {256, 292, 328, 280};
    const int WNP[4] = {256, 304, 336, 288};
    const float* emb = (const float*)d_in[41];
    const int* z     = (const int*)d_in[42];
    const float* pos = (const float*)d_in[43];
    const int* batch = (const int*)d_in[44];
    const int* src   = (const int*)d_in[45];
    const int* dst   = (const int*)d_in[46];

    // --- workspace carve-up
    char* base = (char*)d_ws;
    size_t off = 0;
    auto A = [&](size_t b) -> char* { char* r = base + off; off = (off + b + 255) & ~(size_t)255; return r; };
    unsigned short* radbf = (unsigned short*)A((size_t)NE * 64 * 2);
    float* cbuf = (float*)A((size_t)NE * 4);
    float* shv  = (float*)A((size_t)NE * 12);
    unsigned short* W0b = (unsigned short*)A(256 * 64 * 2);
    unsigned short* W1b = (unsigned short*)A(256 * 256 * 2);
    unsigned short* W2b = (unsigned short*)A(336 * 256 * 2);
    float* h_s   = (float*)A((size_t)NN * 128 * 4);
    float* h_sm  = (float*)A((size_t)NN * 128 * 4);
    float* h_vm  = (float*)A((size_t)NN * 36 * 4);
    float* h_vp  = (float*)A((size_t)NN * 36 * 4);
    float* x1_s  = (float*)A((size_t)NN * 128 * 4);
    float* x1_sm = (float*)A((size_t)NN * 128 * 4);
    float* x1_vm = (float*)A((size_t)NN * 36 * 4);
    float* x1_vp = (float*)A((size_t)NN * 36 * 4);
    float* out_s  = (float*)A((size_t)NN * 152 * 4);
    float* out_sm = (float*)A((size_t)NN * 128 * 4);
    float* out_vm = (float*)A((size_t)NN * 36 * 4);
    float* out_vp = (float*)A((size_t)NN * 36 * 4);
    float* ag_s  = (float*)A((size_t)NN * 140 * 4);
    float* ag_sm = (float*)A((size_t)NN * 140 * 4);
    float* ag_vm = (float*)A((size_t)NN * 152 * 3 * 4);
    float* ag_vp = (float*)A((size_t)NN * 24 * 3 * 4);
    // remaining -> per-edge weight chunk (kept L2-resident)
    size_t rem = (ws_size > off) ? (ws_size - off) : 0;
    int chunk = (int)(rem / ((size_t)336 * 4));
    chunk &= ~63;
    if (chunk > 24576) chunk = 24576;
    if (chunk < 64) chunk = 64;
    float* wbuf = (float*)(base + off);

    auto zero = [&](float* p, size_t n) { k_zero<<<cdiv((int)n, 256), 256, 0, stream>>>(p, (int)n); };
    auto lin = [&](const void* W, const float* x, float* y, int mi, int mo, int nc, int acc) {
        int tot = NN * mo * nc;
        k_linear<<<cdiv(tot, 256), 256, 0, stream>>>((const float*)W, x, y, mi, mo, nc,
                                                     1.0f / sqrtf((float)mi), acc);
    };
    auto conv_w = [&](int L) {
        k_w2bf<<<cdiv(256 * 64, 256), 256, 0, stream>>>(FC0[L], W0b, 256, 50, 64, 256);
        k_w2bf<<<cdiv(256 * 256, 256), 256, 0, stream>>>(FC1[L], W1b, 256, 256, 256, 256);
        k_w2bf<<<cdiv(WNP[L] * 256, 256), 256, 0, stream>>>(FC2[L], W2b, WN[L], 256, 256, WNP[L]);
    };

    // --- geometry + embedding
    k_edge_geom<<<cdiv(NE, 128), 128, 0, stream>>>(pos, src, dst, radbf, cbuf, shv);
    k_emb<<<cdiv(NN * 128, 256), 256, 0, stream>>>(emb, z, h_s);

    // ======================= Layer 0 =======================
    conv_w(0);
    lin(d_in[3], h_s, x1_s, 128, 128, 1, 0);       // lin1 '0,1'
    lin(d_in[6], h_s, out_s, 128, 140, 1, 0);      // si '0,1'
    zero(out_vm, (size_t)NN * 36);
    zero(ag_s, (size_t)NN * 128);
    zero(ag_vm, (size_t)NN * 128 * 3);
    for (int eb = 0; eb < NE; eb += chunk) {
        int ec = NE - eb; if (ec > chunk) ec = chunk;
        k_radial_mlp<<<cdiv(ec, 64), 128, 0, stream>>>(radbf, cbuf, W0b, W1b, W2b, wbuf, eb, ec, WNP[0]);
        k_tp0<<<cdiv(ec * 128, 256), 256, 0, stream>>>(wbuf, WNP[0], x1_s, src, dst, shv, ag_s, ag_vm, eb, ec);
    }
    lin(d_in[4], ag_s, out_s, 128, 140, 1, 1);     // lin2 '0,1'
    lin(d_in[5], ag_vm, out_vm, 128, 12, 3, 1);    // lin2 '1,-1'
    k_gate_s<<<cdiv(NN * 128, 256), 256, 0, stream>>>(out_s, 140, h_s);
    k_gatev<<<cdiv(NN * 36, 256), 256, 0, stream>>>(out_s, 140, 128, out_vm, h_vm);

    // ======================= Layer 1 =======================
    conv_w(1);
    lin(d_in[10], h_s, x1_s, 128, 128, 1, 0);
    lin(d_in[11], h_vm, x1_vm, 12, 12, 3, 0);
    lin(d_in[15], h_s, out_s, 128, 152, 1, 0);
    lin(d_in[16], h_vm, out_vm, 12, 12, 3, 0);
    zero(out_vp, (size_t)NN * 36);
    zero(ag_s, (size_t)NN * 140);
    zero(ag_vm, (size_t)NN * 140 * 3);
    zero(ag_vp, (size_t)NN * 12 * 3);
    for (int eb = 0; eb < NE; eb += chunk) {
        int ec = NE - eb; if (ec > chunk) ec = chunk;
        k_radial_mlp<<<cdiv(ec, 64), 128, 0, stream>>>(radbf, cbuf, W0b, W1b, W2b, wbuf, eb, ec, WNP[1]);
        k_tp1<<<cdiv(ec * 128, 256), 256, 0, stream>>>(wbuf, WNP[1], x1_s, x1_vm, src, dst, shv,
                                                       ag_s, ag_vm, ag_vp, eb, ec);
    }
    lin(d_in[12], ag_s, out_s, 140, 152, 1, 1);
    lin(d_in[13], ag_vm, out_vm, 140, 12, 3, 1);
    lin(d_in[14], ag_vp, out_vp, 12, 12, 3, 1);
    k_gate_s<<<cdiv(NN * 128, 256), 256, 0, stream>>>(out_s, 152, h_s);
    k_gatev<<<cdiv(NN * 36, 256), 256, 0, stream>>>(out_s, 152, 128, out_vm, h_vm);
    k_gatev<<<cdiv(NN * 36, 256), 256, 0, stream>>>(out_s, 152, 140, out_vp, h_vp);

    // ======================= Layer 2 =======================
    conv_w(2);
    lin(d_in[20], h_s, x1_s, 128, 128, 1, 0);
    lin(d_in[21], h_vm, x1_vm, 12, 12, 3, 0);
    lin(d_in[22], h_vp, x1_vp, 12, 12, 3, 0);
    lin(d_in[27], h_s, out_s, 128, 152, 1, 0);
    lin(d_in[28], h_vm, out_vm, 12, 12, 3, 0);
    lin(d_in[29], h_vp, out_vp, 12, 12, 3, 0);
    zero(out_sm, (size_t)NN * 128);
    zero(ag_s, (size_t)NN * 140);
    zero(ag_sm, (size_t)NN * 12);
    zero(ag_vm, (size_t)NN * 152 * 3);
    zero(ag_vp, (size_t)NN * 24 * 3);
    for (int eb = 0; eb < NE; eb += chunk) {
        int ec = NE - eb; if (ec > chunk) ec = chunk;
        k_radial_mlp<<<cdiv(ec, 64), 128, 0, stream>>>(radbf, cbuf, W0b, W1b, W2b, wbuf, eb, ec, WNP[2]);
        k_tp2<<<cdiv(ec * 128, 256), 256, 0, stream>>>(wbuf, WNP[2], x1_s, x1_vm, x1_vp, src, dst, shv,
                                                       ag_s, ag_sm, ag_vm, ag_vp, eb, ec);
    }
    lin(d_in[23], ag_sm, out_sm, 12, 128, 1, 1);   // lin2 '0,-1'
    lin(d_in[24], ag_s, out_s, 140, 152, 1, 1);    // lin2 '0,1'
    lin(d_in[25], ag_vm, out_vm, 152, 12, 3, 1);   // lin2 '1,-1'
    lin(d_in[26], ag_vp, out_vp, 24, 12, 3, 1);    // lin2 '1,1'
    k_gate_s<<<cdiv(NN * 128, 256), 256, 0, stream>>>(out_s, 152, h_s);
    k_gate_t<<<cdiv(NN * 128, 256), 256, 0, stream>>>(out_sm, 128, h_sm);
    k_gatev<<<cdiv(NN * 36, 256), 256, 0, stream>>>(out_s, 152, 128, out_vm, h_vm);
    k_gatev<<<cdiv(NN * 36, 256), 256, 0, stream>>>(out_s, 152, 140, out_vp, h_vp);

    // ======================= Layer 3 (final conv) =======================
    conv_w(3);
    lin(d_in[33], h_sm, x1_sm, 128, 128, 1, 0);    // lin1 '0,-1'
    lin(d_in[34], h_s, x1_s, 128, 128, 1, 0);      // lin1 '0,1'
    lin(d_in[35], h_vm, x1_vm, 12, 12, 3, 0);
    lin(d_in[36], h_vp, x1_vp, 12, 12, 3, 0);
    lin(d_in[40], h_s, out_s, 128, 1, 1, 0);       // si '0,1'
    lin(d_in[39], h_sm, out_sm, 128, 1, 1, 0);     // si '0,-1'
    zero(ag_s, (size_t)NN * 140);
    zero(ag_sm, (size_t)NN * 140);
    for (int eb = 0; eb < NE; eb += chunk) {
        int ec = NE - eb; if (ec > chunk) ec = chunk;
        k_radial_mlp<<<cdiv(ec, 64), 128, 0, stream>>>(radbf, cbuf, W0b, W1b, W2b, wbuf, eb, ec, WNP[3]);
        k_tp3<<<cdiv(ec * 128, 256), 256, 0, stream>>>(wbuf, WNP[3], x1_s, x1_sm, x1_vm, x1_vp,
                                                       src, dst, shv, ag_s, ag_sm, eb, ec);
    }
    lin(d_in[38], ag_s, out_s, 140, 1, 1, 1);      // lin2 '0,1'
    lin(d_in[37], ag_sm, out_sm, 140, 1, 1, 1);    // lin2 '0,-1'

    // readout
    zero((float*)d_out, 8);
    k_readout<<<cdiv(NN, 256), 256, 0, stream>>>(out_s, out_sm, batch, (float*)d_out);
}
